// GPT_8392366097166
// MI455X (gfx1250) — matrix operand order repaired and checked by reading, not compiled, not run on hardware
//
#include <hip/hip_runtime.h>
#include <hip/hip_bf16.h>
#include <math.h>

typedef __bf16 bf16;
typedef __attribute__((ext_vector_type(16))) __bf16 v16bf;
typedef __attribute__((ext_vector_type(8)))  float  v8f;
typedef __attribute__((ext_vector_type(4)))  unsigned int u32x4;
typedef __attribute__((ext_vector_type(8)))  int i32x8;
typedef __attribute__((ext_vector_type(4)))  int i32x4;

static constexpr int B_  = 8;
static constexpr int T_  = 2048;
static constexpr int E_  = 256;
static constexpr int H_  = 2;
static constexpr int HD_ = 128;   // E/H
static constexpr int M2_ = 134;
static constexpr float SCALE_ = 0.08838834764831845f; // 1/sqrt(128)

// ---------------- workspace layout (bytes) ----------------
static constexpr size_t OFF_X     = 0;                         // B*T*E f32   = 16 MB
static constexpr size_t OFF_H     = 16777216;                  // B*T*E bf16  = 8 MB
static constexpr size_t OFF_Q     = 25165824;                  // B*H*T*HD bf16 = 8 MB
static constexpr size_t OFF_K     = 33554432;                  // 8 MB
static constexpr size_t OFF_VT    = 41943040;                  // v transposed (B,H,HD,T) 8 MB
static constexpr size_t OFF_Y     = 50331648;                  // attn out (B*T,E) bf16 8 MB
static constexpr size_t OFF_AFC   = 25165824;                  // reuse Q+K region: (B*T,512) bf16 16 MB
static constexpr size_t OFF_WATTN = 58720256;                  // 3E*E bf16
static constexpr size_t OFF_WPROJ = 59113472;                  // E*E bf16
static constexpr size_t OFF_WFC   = 59244544;                  // 2E*E bf16
static constexpr size_t OFF_WMLP  = 59506688;                  // E*2E bf16
static constexpr size_t OFF_EMB   = 59768832;                  // B*E f32

// ---------------- WMMA fragment helpers ----------------
// A fragment, 16(M) x 32(K) bf16 from row-major src with leading dim ld.
__device__ __forceinline__ v16bf load_a(const bf16* base, int ld) {
  int lane = threadIdx.x & 31;
  const bf16* p = base + (size_t)(lane & 15) * ld + ((lane >> 4) << 3);
  v16bf f;
#pragma unroll
  for (int i = 0; i < 8; ++i) { f[i] = p[i]; f[i + 8] = p[i + 16]; }
  return f;
}
// B fragment, 32(K) x 16(N) bf16; column n of B is row n of src (row-major, ld).
__device__ __forceinline__ v16bf load_b(const bf16* base, int ld) {
  int lane = threadIdx.x & 31;
  const bf16* p = base + (size_t)(lane & 15) * ld + ((lane >> 4) << 3);
  v16bf f;
#pragma unroll
  for (int i = 0; i < 8; ++i) { f[i] = p[i]; f[i + 8] = p[i + 16]; }
  return f;
}
__device__ __forceinline__ v8f wmma_bf16(v16bf a, v16bf b, v8f c) {
  return __builtin_amdgcn_wmma_f32_16x16x32_bf16(false, a, false, b, (short)0, c, false, false);
}

// ---------------- TDM: 2-D tile Global -> LDS (bf16 elements) ----------------
// Builds D# per CDNA5 ISA ch.8: group0 {count/lds/global/type}, group1 {dims}.
// 6-arg builtin form (clang-23 toolchain): (g0 v4u, g1 v8i, g2 v4i, g3 v4i, v8i, cpol)
__device__ __forceinline__ void tdm_load_2d(unsigned lds_addr, const bf16* gptr,
                                            unsigned tile_w, unsigned tile_h,
                                            unsigned stride_elems) {
  unsigned long long ga = (unsigned long long)(uintptr_t)gptr;
  u32x4 g0;
  g0[0] = 1u;                                                  // count=1, user descriptor
  g0[1] = lds_addr;                                            // LDS byte address
  g0[2] = (unsigned)ga;                                        // global_addr[31:0]
  g0[3] = (unsigned)((ga >> 32) & 0x01FFFFFFu) | (2u << 30);   // global_addr[56:32], type=2
  i32x8 g1;
  g1[0] = 0x00010000;                                          // mask=0, data_size=1 (2 bytes)
  g1[1] = (int)((tile_w & 0xFFFFu) << 16);                     // tensor_dim0[15:0]
  g1[2] = (int)(((tile_w >> 16) & 0xFFFFu) |                   // tensor_dim0[31:16]
                ((tile_h & 0xFFFFu) << 16));                   // tensor_dim1[15:0]
  g1[3] = (int)(((tile_h >> 16) & 0xFFFFu) |                   // tensor_dim1[31:16]
                ((tile_w & 0xFFFFu) << 16));                   // tile_dim0
  g1[4] = (int)(tile_h & 0xFFFFu);                             // tile_dim1; tile_dim2=0
  g1[5] = (int)stride_elems;                                   // tensor_dim0_stride[31:0]
  g1[6] = 0;
  g1[7] = 0;
  i32x4 gz4 = {0, 0, 0, 0};
  i32x8 gz8 = {0, 0, 0, 0, 0, 0, 0, 0};
  __builtin_amdgcn_tensor_load_to_lds(g0, g1, gz4, gz4, gz8, 0);
}

// ---------------- elementwise / LN kernels ----------------
__global__ void f32_to_bf16_kernel(const float* __restrict__ s, bf16* __restrict__ d, int n) {
  int i = blockIdx.x * 256 + threadIdx.x;
  if (i < n) d[i] = (bf16)s[i];
}

// x = relu(X + wpe[t] + wpe[cls]);  h = LN(x)*g + b   (one block = one row of E=256)
__global__ void embed_ln1_kernel(const float* __restrict__ X, const float* __restrict__ wpe,
                                 const float* __restrict__ g, const float* __restrict__ bt,
                                 float* __restrict__ x, bf16* __restrict__ h) {
  int row = blockIdx.x;          // b*T + t
  int t = row & (T_ - 1);
  int e = threadIdx.x;
  int cls = (t < T_ / 2) ? 0 : 1;
  float v = X[(size_t)row * E_ + e] + wpe[(size_t)t * E_ + e] + wpe[(size_t)cls * E_ + e];
  v = fmaxf(v, 0.f);
  x[(size_t)row * E_ + e] = v;
  __shared__ float red[256];
  red[e] = v; __syncthreads();
  for (int s = 128; s > 0; s >>= 1) { if (e < s) red[e] += red[e + s]; __syncthreads(); }
  float mu = red[0] * (1.f / E_);
  __syncthreads();
  float d = v - mu;
  red[e] = d * d; __syncthreads();
  for (int s = 128; s > 0; s >>= 1) { if (e < s) red[e] += red[e + s]; __syncthreads(); }
  float var = red[0] * (1.f / E_);
  h[(size_t)row * E_ + e] = (bf16)(d * rsqrtf(var + 1e-5f) * g[e] + bt[e]);
}

__global__ void ln2_kernel(const float* __restrict__ x, const float* __restrict__ g,
                           const float* __restrict__ bt, bf16* __restrict__ h) {
  int row = blockIdx.x; int e = threadIdx.x;
  float v = x[(size_t)row * E_ + e];
  __shared__ float red[256];
  red[e] = v; __syncthreads();
  for (int s = 128; s > 0; s >>= 1) { if (e < s) red[e] += red[e + s]; __syncthreads(); }
  float mu = red[0] * (1.f / E_);
  __syncthreads();
  float d = v - mu;
  red[e] = d * d; __syncthreads();
  for (int s = 128; s > 0; s >>= 1) { if (e < s) red[e] += red[e + s]; __syncthreads(); }
  float var = red[0] * (1.f / E_);
  h[(size_t)row * E_ + e] = (bf16)(d * rsqrtf(var + 1e-5f) * g[e] + bt[e]);
}

// x = relu( LN(relu(x)) * g + b )   in-place
__global__ void lnf_kernel(float* __restrict__ x, const float* __restrict__ g,
                           const float* __restrict__ bt) {
  int row = blockIdx.x; int e = threadIdx.x;
  float v = fmaxf(x[(size_t)row * E_ + e], 0.f);
  __shared__ float red[256];
  red[e] = v; __syncthreads();
  for (int s = 128; s > 0; s >>= 1) { if (e < s) red[e] += red[e + s]; __syncthreads(); }
  float mu = red[0] * (1.f / E_);
  __syncthreads();
  float d = v - mu;
  red[e] = d * d; __syncthreads();
  for (int s = 128; s > 0; s >>= 1) { if (e < s) red[e] += red[e + s]; __syncthreads(); }
  float var = red[0] * (1.f / E_);
  x[(size_t)row * E_ + e] = fmaxf(d * rsqrtf(var + 1e-5f) * g[e] + bt[e], 0.f);
}

// ---------------- generic WMMA GEMM:  out = epi(A @ W^T + bias) ----------------
// A: (M,K) bf16 row-major. W: (N,K) bf16 row-major. One wave -> 16x64 tile.
// mode 0: QKV scatter (relu; q,k row-major per head; v transposed)
// mode 1: residual: xres[m*E + n] += val
// mode 2: relu -> o0[m*N + n] bf16
__global__ void gemm_bf16_kernel(const bf16* __restrict__ A, const bf16* __restrict__ W,
                                 const float* __restrict__ bias,
                                 int Mtiles, int Ntiles64, int K, int mode,
                                 float* __restrict__ xres,
                                 bf16* __restrict__ o0, bf16* __restrict__ o1,
                                 bf16* __restrict__ o2) {
  int wave = threadIdx.x >> 5;
  int lane = threadIdx.x & 31;
  int tile = blockIdx.x * 8 + wave;
  if (tile >= Mtiles * Ntiles64) return;
  int tm = tile / Ntiles64, tn = tile % Ntiles64;
  const bf16* Ab = A + (size_t)tm * 16 * K;
  v8f acc[4] = {};
  for (int k0 = 0; k0 < K; k0 += 32) {
    v16bf a = load_a(Ab + k0, K);
#pragma unroll
    for (int j = 0; j < 4; ++j) {
      v16bf b = load_b(W + (size_t)(tn * 64 + j * 16) * K + k0, K);
      acc[j] = wmma_bf16(a, b, acc[j]);
    }
  }
  int N = Ntiles64 * 64;
  int mbase = tm * 16 + ((lane >> 4) << 3);
  int nlo = lane & 15;
#pragma unroll
  for (int j = 0; j < 4; ++j) {
#pragma unroll
    for (int r = 0; r < 8; ++r) {
      int m = mbase + r;
      int n = tn * 64 + j * 16 + nlo;
      float val = acc[j][r] + bias[n];
      if (mode == 0) {
        val = fmaxf(val, 0.f);
        int b = m >> 11, t = m & (T_ - 1);
        int part = n >> 8, c = n & 255, hh = c >> 7, d = c & 127;
        if (part == 2) {
          o2[((size_t)(b * H_ + hh) * HD_ + d) * T_ + t] = (bf16)val;   // vT
        } else {
          size_t idx = ((size_t)(b * H_ + hh) * T_ + t) * HD_ + d;
          if (part == 0) o0[idx] = (bf16)val; else o1[idx] = (bf16)val;
        }
      } else if (mode == 1) {
        xres[(size_t)m * E_ + n] += val;
      } else {
        o0[(size_t)m * N + n] = (bf16)fmaxf(val, 0.f);
      }
    }
  }
}

// ---------------- causal relu-attention with TDM-staged K/V ----------------
// Block = 8 waves = 128 q rows of one (b,h). K/V streamed chunk-by-chunk into
// LDS by the Tensor Data Mover (double buffered, TENSORcnt-tracked); all 8
// waves consume each chunk from LDS (8x reuse vs. global streaming).
__global__ __launch_bounds__(256) void attn_kernel(const bf16* __restrict__ q,
                                                   const bf16* __restrict__ k,
                                                   const bf16* __restrict__ vT,
                                                   bf16* __restrict__ y) {
  __shared__ bf16 kbuf[2][32 * HD_];     // 2 x 8 KB : K rows k0..k0+31, ld=128
  __shared__ bf16 vbuf[2][HD_ * 32];     // 2 x 8 KB : vT rows d=0..127, ld=32
  __shared__ bf16 stile[8][16 * 32];     // per-wave S staging (D->A relayout)
  int wave = threadIdx.x >> 5;
  int lane = threadIdx.x & 31;
  int bh = blockIdx.y;
  int b = bh / H_, hh = bh % H_;
  int qt = blockIdx.x * 8 + wave;
  const bf16* qp = q + ((size_t)bh * T_ + (size_t)qt * 16) * HD_;
  const bf16* kp = k + (size_t)bh * T_ * HD_;
  const bf16* vp = vT + (size_t)bh * HD_ * T_;
  v16bf qf[4];
#pragma unroll
  for (int s = 0; s < 4; ++s) qf[s] = load_a(qp + s * 32, HD_);
  v8f acc[8] = {};
  int mrow = (lane >> 4) << 3;
  int ncol = lane & 15;
  int kmax_w = qt * 16 + 16;                 // this wave's causal bound
  int nch = blockIdx.x * 4 + 4;              // 32-wide chunks for whole block
  bool issuer = (threadIdx.x < 32);          // wave 0 drives the TDM
  if (issuer) {
    tdm_load_2d((unsigned)(uintptr_t)&kbuf[0][0], kp,       HD_, 32, HD_);
    tdm_load_2d((unsigned)(uintptr_t)&vbuf[0][0], vp,       32, HD_, T_);
  }
  for (int c = 0; c < nch; ++c) {
    int k0 = c * 32;
    if (issuer) {
      if (c + 1 < nch) {
        int k1 = (c + 1) * 32;
        tdm_load_2d((unsigned)(uintptr_t)&kbuf[(c + 1) & 1][0], kp + (size_t)k1 * HD_, HD_, 32, HD_);
        tdm_load_2d((unsigned)(uintptr_t)&vbuf[(c + 1) & 1][0], vp + k1,               32, HD_, T_);
        __builtin_amdgcn_s_wait_tensorcnt(2);   // chunk c landed, c+1 in flight
      } else {
        __builtin_amdgcn_s_wait_tensorcnt(0);
      }
    }
    __syncthreads();                            // chunk c visible to all waves
    if (k0 < kmax_w) {                          // wave-uniform causal skip
      const bf16* kb = &kbuf[c & 1][0];
      const bf16* vb = &vbuf[c & 1][0];
      v8f s0 = {}, s1 = {};
#pragma unroll
      for (int s = 0; s < 4; ++s) {
        v16bf b0 = load_b(kb + s * 32, HD_);             // k rows k0..k0+15
        s0 = wmma_bf16(qf[s], b0, s0);
        v16bf b1 = load_b(kb + 16 * HD_ + s * 32, HD_);  // k rows k0+16..k0+31
        s1 = wmma_bf16(qf[s], b1, s1);
      }
      bf16* st = &stile[wave][0];
#pragma unroll
      for (int r = 0; r < 8; ++r) {
        int mm = mrow + r;
        int qg = qt * 16 + mm;
        float v0 = (k0 + ncol      <= qg) ? fmaxf(s0[r] * SCALE_, 0.f) : 0.f;
        float v1 = (k0 + 16 + ncol <= qg) ? fmaxf(s1[r] * SCALE_, 0.f) : 0.f;
        st[mm * 32 + ncol]      = (bf16)v0;
        st[mm * 32 + 16 + ncol] = (bf16)v1;
      }
      // per-wave LDS roundtrip: DS ops are in-order within a wave
      v16bf af = load_a(st, 32);
#pragma unroll
      for (int j = 0; j < 8; ++j) {
        v16bf bv = load_b(vb + (size_t)(j * 16) * 32, 32);
        acc[j] = wmma_bf16(af, bv, acc[j]);
      }
    }
    __syncthreads();                            // release buffer (c&1) for reuse
  }
#pragma unroll
  for (int j = 0; j < 8; ++j) {
#pragma unroll
    for (int r = 0; r < 8; ++r) {
      int t = qt * 16 + mrow + r;
      int d = j * 16 + ncol;
      y[((size_t)b * T_ + t) * E_ + hh * HD_ + d] = (bf16)acc[j][r];
    }
  }
}

// ---------------- pooling + head + losses ----------------
__global__ void emb_mean_kernel(const float* __restrict__ x, float* __restrict__ emb) {
  int b = blockIdx.x; int e = threadIdx.x;
  const float* p = x + (size_t)b * T_ * E_ + e;
  float s = 0.f;
  for (int t = 0; t < T_; ++t) s += p[(size_t)t * E_];
  emb[b * E_ + e] = s * (1.f / T_);
}

__global__ void final_kernel(const float* __restrict__ emb, const float* __restrict__ Y,
                             const float* __restrict__ head_w, const float* __restrict__ head_b,
                             float* __restrict__ out) {
  __shared__ float s_emb[B_ * E_];
  __shared__ float s_log[B_ * M2_];
  __shared__ float red[256];
  int tid = threadIdx.x;
  for (int i = tid; i < B_ * E_; i += 256) s_emb[i] = emb[i];
  __syncthreads();
  for (int i = tid; i < B_ * M2_; i += 256) {
    int b = i / M2_, n = i % M2_;
    float a = head_b[n];
    for (int e = 0; e < E_; ++e) a += s_emb[b * E_ + e] * head_w[(size_t)n * E_ + e];
    a = fmaxf(a, 0.f);
    s_log[i] = a;
    out[2051 + i] = a;                       // logits
  }
  __syncthreads();
  float l1 = 0.f;
  for (int i = tid; i < B_ * M2_; i += 256) { float d = s_log[i] - Y[i]; l1 += d * d; }
  red[tid] = l1; __syncthreads();
  for (int s = 128; s > 0; s >>= 1) { if (tid < s) red[tid] += red[tid + s]; __syncthreads(); }
  float loss1 = sqrtf(red[0] / (float)(B_ * M2_));
  __syncthreads();
  float l3 = 0.f;
  for (int i = tid; i < B_ * (E_ / 2); i += 256) {
    int b = i / (E_ / 2), j = i % (E_ / 2);
    float a = s_emb[b * E_ + j], c = s_emb[b * E_ + (E_ / 2) + j];
    out[i] = a;                              // e1
    out[1024 + i] = c;                       // e2
    float d = a - c; l3 += d * d;
  }
  red[tid] = l3; __syncthreads();
  for (int s = 128; s > 0; s >>= 1) { if (tid < s) red[tid] += red[tid + s]; __syncthreads(); }
  float loss3 = sqrtf(red[0] / (float)(B_ * (E_ / 2)));
  if (tid == 0) {
    out[2048] = 50.f * loss1 + loss3;
    out[2049] = loss1;
    out[2050] = loss3;
  }
}

// ---------------- host launcher ----------------
extern "C" void kernel_launch(void* const* d_in, const int* in_sizes, int n_in,
                              void* d_out, int out_size, void* d_ws, size_t ws_size,
                              hipStream_t stream) {
  const float* X          = (const float*)d_in[0];
  const float* Y          = (const float*)d_in[1];
  const float* wpe        = (const float*)d_in[2];
  const float* ln1_g      = (const float*)d_in[3];
  const float* ln1_b      = (const float*)d_in[4];
  const float* attn_w     = (const float*)d_in[5];
  const float* attn_b     = (const float*)d_in[6];
  const float* attnproj_w = (const float*)d_in[7];
  const float* attnproj_b = (const float*)d_in[8];
  const float* ln2_g      = (const float*)d_in[9];
  const float* ln2_b      = (const float*)d_in[10];
  const float* fc_w       = (const float*)d_in[11];
  const float* fc_b       = (const float*)d_in[12];
  const float* mlpproj_w  = (const float*)d_in[13];
  const float* mlpproj_b  = (const float*)d_in[14];
  const float* lnf_g      = (const float*)d_in[15];
  const float* lnf_b      = (const float*)d_in[16];
  const float* head_w     = (const float*)d_in[17];
  const float* head_b     = (const float*)d_in[18];

  char* ws = (char*)d_ws;
  float* x      = (float*)(ws + OFF_X);
  bf16*  h      = (bf16*)(ws + OFF_H);
  bf16*  qb     = (bf16*)(ws + OFF_Q);
  bf16*  kb     = (bf16*)(ws + OFF_K);
  bf16*  vTb    = (bf16*)(ws + OFF_VT);
  bf16*  yb     = (bf16*)(ws + OFF_Y);
  bf16*  afc    = (bf16*)(ws + OFF_AFC);
  bf16*  w_attn = (bf16*)(ws + OFF_WATTN);
  bf16*  w_proj = (bf16*)(ws + OFF_WPROJ);
  bf16*  w_fc   = (bf16*)(ws + OFF_WFC);
  bf16*  w_mlp  = (bf16*)(ws + OFF_WMLP);
  float* emb    = (float*)(ws + OFF_EMB);

  // weight fp32 -> bf16
  f32_to_bf16_kernel<<<(3*E_*E_ + 255)/256, 256, 0, stream>>>(attn_w,     w_attn, 3*E_*E_);
  f32_to_bf16_kernel<<<(E_*E_   + 255)/256, 256, 0, stream>>>(attnproj_w, w_proj, E_*E_);
  f32_to_bf16_kernel<<<(2*E_*E_ + 255)/256, 256, 0, stream>>>(fc_w,       w_fc,   2*E_*E_);
  f32_to_bf16_kernel<<<(E_*2*E_ + 255)/256, 256, 0, stream>>>(mlpproj_w,  w_mlp,  E_*2*E_);

  int rows = B_ * T_;          // 16384
  int Mtiles = rows / 16;      // 1024

  // embed + relu + LN1
  embed_ln1_kernel<<<rows, 256, 0, stream>>>(X, wpe, ln1_g, ln1_b, x, h);

  // QKV: (rows,256) @ (768,256)^T ; scatter q,k,vT with relu
  {
    int Ntiles = 768 / 64, total = Mtiles * Ntiles;
    gemm_bf16_kernel<<<(total + 7)/8, 256, 0, stream>>>(h, w_attn, attn_b, Mtiles, Ntiles, E_,
                                                        0, nullptr, qb, kb, vTb);
  }

  // attention (TDM-staged K/V, 8 q-tiles per block)
  attn_kernel<<<dim3(T_/128, B_*H_), 256, 0, stream>>>(qb, kb, vTb, yb);

  // x += y @ attnproj^T + b
  {
    int Ntiles = E_ / 64, total = Mtiles * Ntiles;
    gemm_bf16_kernel<<<(total + 7)/8, 256, 0, stream>>>(yb, w_proj, attnproj_b, Mtiles, Ntiles, E_,
                                                        1, x, nullptr, nullptr, nullptr);
  }

  // LN2
  ln2_kernel<<<rows, 256, 0, stream>>>(x, ln2_g, ln2_b, h);

  // a = relu(h @ fc^T + b)  -> (rows, 512) bf16
  {
    int Ntiles = (2*E_) / 64, total = Mtiles * Ntiles;
    gemm_bf16_kernel<<<(total + 7)/8, 256, 0, stream>>>(h, w_fc, fc_b, Mtiles, Ntiles, E_,
                                                        2, nullptr, afc, nullptr, nullptr);
  }

  // x += a @ mlpproj^T + b   (K = 512)
  {
    int Ntiles = E_ / 64, total = Mtiles * Ntiles;
    gemm_bf16_kernel<<<(total + 7)/8, 256, 0, stream>>>(afc, w_mlp, mlpproj_b, Mtiles, Ntiles, 2*E_,
                                                        1, x, nullptr, nullptr, nullptr);
  }

  // x = relu(LN(relu(x)))
  lnf_kernel<<<rows, 256, 0, stream>>>(x, lnf_g, lnf_b);

  // emb = mean over T
  emb_mean_kernel<<<B_, 256, 0, stream>>>(x, emb);

  // head + losses + outputs
  final_kernel<<<1, 256, 0, stream>>>(emb, Y, head_w, head_b, (float*)d_out);
}
